// MiddleTransformer_12541304504992
// MI455X (gfx1250) — matrix operand order repaired and checked by reading, not compiled, not run on hardware
//
#include <hip/hip_runtime.h>
#include <hip/hip_bf16.h>
#include <math.h>

#define DEVI __device__ __forceinline__

typedef __bf16 bf16;
typedef __attribute__((ext_vector_type(16))) __bf16 v16bf;
typedef __attribute__((ext_vector_type(8)))  __bf16 v8bf;
typedef __attribute__((ext_vector_type(8)))  float  v8f;
typedef __attribute__((ext_vector_type(4)))  unsigned int u32x4;

// ---- static dims ----
#define TD   512      // model dim T
#define FDIM 2048     // ffn dim
#define NHD  8        // heads
#define BSZ  32       // batch
#define HSZ  64
#define WSZ  128
#define LMAX 129

DEVI v8f vzero8() { v8f z = {0.f,0.f,0.f,0.f,0.f,0.f,0.f,0.f}; return z; }
DEVI float mishf(float x) { float sp = (x > 20.f) ? x : log1pf(expf(x)); return x * tanhf(sp); }
DEVI float geluf(float x) { return 0.5f * x * (1.f + erff(x * 0.70710678118654752f)); }
DEVI float actf(float x, int act) { return act == 1 ? mishf(x) : (act == 2 ? geluf(x) : x); }

// Low 32 bits of a generic pointer to LDS = LDS byte offset (ISA: flat->LDS truncation).
DEVI unsigned lds_off(const void* p) { return (unsigned)(unsigned long long)p; }

// CDNA5 async DMA: global memory -> LDS, 16B per lane, tracked by ASYNCcnt.
DEVI void async_b128(unsigned lds, const bf16* g) {
  asm volatile("global_load_async_to_lds_b128 %0, %1, off" :: "v"(lds), "v"(g) : "memory");
}

// A-matrix fragment (16xK tile, interleaved K pattern per ISA 7.12.2):
// lane L: row = L%16; elems 0..7 -> K = half*8 + e ; elems 8..15 -> K = 16 + half*8 + e
DEVI v16bf frag_a_rm(const bf16* base, int ld) {
  const int lane = threadIdx.x & 31;
  const bf16* p = base + (lane & 15) * ld + ((lane >> 4) << 3);
  v8bf lo = *(const v8bf*)(p);
  v8bf hi = *(const v8bf*)(p + 16);
  v16bf f;
#pragma unroll
  for (int j = 0; j < 8; ++j) { f[j] = lo[j]; f[j + 8] = hi[j]; }
  return f;
}

// B-matrix fragment from row-major W[N][K] rows (interleaved K, same as A):
// lane L: n = L%16; elems 0..7 -> K = half*8 + e ; elems 8..15 -> K = 16 + half*8 + e
DEVI v16bf frag_b_rm(const bf16* base, int ld) {
  const int lane = threadIdx.x & 31;
  const bf16* p = base + (lane & 15) * ld + ((lane >> 4) << 3);
  v8bf lo = *(const v8bf*)(p);
  v8bf hi = *(const v8bf*)(p + 16);
  v16bf f;
#pragma unroll
  for (int j = 0; j < 8; ++j) { f[j] = lo[j]; f[j + 8] = hi[j]; }
  return f;
}

// B-matrix fragment gathered from row-major storage [K][N] (column per lane)
DEVI v16bf frag_b_cm(const bf16* base, int ld) {
  const int lane = threadIdx.x & 31;
  const int n = lane & 15, h16 = (lane >> 4) << 4;
  v16bf f;
#pragma unroll
  for (int e = 0; e < 16; ++e) f[e] = base[((h16 >> 1) + e + (e & 8)) * ld + n];
  return f;
}

DEVI v8f wmma_bf(v16bf a, v16bf b, v8f c) {
  return __builtin_amdgcn_wmma_f32_16x16x32_bf16(false, a, false, b, (short)0, c, false, false);
}

// ======================= generic GEMM: Y = act(A @ W^T + bias) =======================
// A: bf16 [M,K] row-major; W: bf16 [N,K] row-major; bias f32[N] or null.
// Yf (f32) and/or Yb (bf16) outputs [M,N].
// Double-buffered LDS tiles filled by GLOBAL_LOAD_ASYNC_TO_LDS_B128 (ASYNCcnt),
// so the next K-chunk streams in while WMMAs consume the current one.
#define GBM 64
#define GBN 128
#define GLDA 40

__global__ __launch_bounds__(256)
void gemm_kernel(const bf16* __restrict__ A, const bf16* __restrict__ W,
                 const float* __restrict__ bias, float* __restrict__ Yf,
                 bf16* __restrict__ Yb, int M, int N, int K, int act)
{
  __shared__ __align__(16) bf16 As[2][GBM * GLDA];
  __shared__ __align__(16) bf16 Ws[2][GBN * GLDA];
  const int tid = threadIdx.x;
  const int lane = tid & 31, wave = tid >> 5;
  const int wm = wave & 3, wn = wave >> 2;
  const int m0 = blockIdx.y * GBM, n0 = blockIdx.x * GBN;

  v8f acc[4];
#pragma unroll
  for (int f = 0; f < 4; ++f) acc[f] = vzero8();

  const int ar = tid >> 2, as = (tid & 3) * 8;    // A stage: 64 rows x 4 segs of 8
  const int wr = tid >> 1, wsg = (tid & 1) * 16;  // W stage: 128 rows x 2 segs of 16
  const int am = (m0 + ar < M) ? (m0 + ar) : (M - 1);
  const int wn0 = (n0 + wr < N) ? (n0 + wr) : (N - 1);
  const bf16* Ap = A + (long)am * K + as;
  const bf16* Wp = W + (long)wn0 * K + wsg;

  // per-thread LDS destination addresses for the async DMA, both buffers
  const unsigned ldsA0 = lds_off(&As[0][ar * GLDA + as]);
  const unsigned ldsA1 = lds_off(&As[1][ar * GLDA + as]);
  const unsigned ldsW0 = lds_off(&Ws[0][wr * GLDA + wsg]);
  const unsigned ldsW1 = lds_off(&Ws[1][wr * GLDA + wsg]);

  // prologue: start chunk 0 into buffer 0
  async_b128(ldsA0, Ap);
  async_b128(ldsW0, Wp);
  async_b128(ldsW0 + 16, Wp + 8);

  int cur = 0;
  for (int kt = 0; kt < K; kt += 32) {
    __syncthreads();  // all waves done reading buffer cur^1 (next async target)
    if (kt + 32 < K) {
      const unsigned nA = cur ? ldsA0 : ldsA1;
      const unsigned nW = cur ? ldsW0 : ldsW1;
      async_b128(nA, Ap + kt + 32);
      async_b128(nW, Wp + kt + 32);
      async_b128(nW + 16, Wp + kt + 40);
      asm volatile("s_wait_asynccnt 0x3" ::: "memory");  // chunk `cur` landed
    } else {
      asm volatile("s_wait_asynccnt 0x0" ::: "memory");
    }
    __syncthreads();  // all waves observed their async completions

    const bf16* Ab = As[cur];
    const bf16* Wb = Ws[cur];
    v16bf a = frag_a_rm(Ab + (wm * 16) * GLDA, GLDA);
#pragma unroll
    for (int f = 0; f < 4; ++f) {
      v16bf b = frag_b_rm(Wb + (wn * 64 + f * 16) * GLDA, GLDA);
      acc[f] = wmma_bf(a, b, acc[f]);
    }
    cur ^= 1;
  }
  const int cn = lane & 15, rh = (lane >> 4) * 8;
#pragma unroll
  for (int f = 0; f < 4; ++f) {
    int n = n0 + wn * 64 + f * 16 + cn;
    if (n >= N) continue;
    float bv = bias ? bias[n] : 0.f;
#pragma unroll
    for (int j = 0; j < 8; ++j) {
      int m = m0 + wm * 16 + rh + j;
      if (m >= M) continue;
      float v = actf(acc[f][j] + bv, act);
      if (Yf) Yf[(long)m * N + n] = v;
      if (Yb) Yb[(long)m * N + n] = (bf16)v;
    }
  }
}

// ======================= fused ToTimeSeries =======================
// per block: one (b,w) token column; mish(x@w1^T+b1) -> LDS(bf16) -> WMMA @ w2^T -> mish -> sum over H
#define TLDH 1032
__global__ __launch_bounds__(256)
void tts_kernel(const float* __restrict__ x, const bf16* __restrict__ w1,
                const float* __restrict__ b1, const bf16* __restrict__ w2,
                const float* __restrict__ b2, const float* __restrict__ pe,
                float* __restrict__ srcf, bf16* __restrict__ srcb)
{
  __shared__ __align__(16) bf16 h1s[16 * TLDH];
  __shared__ __align__(16) bf16 w1s[1024 * 8];
  __shared__ float xs[64 * 8];
  const int tid = threadIdx.x, lane = tid & 31, wave = tid >> 5;
  const int bw = blockIdx.x;
  const int b = bw >> 7, w = bw & 127;

  for (int i = tid; i < 64 * 8; i += 256) {
    int h = i >> 3, c = i & 7;
    xs[i] = x[(((b * 8 + c) * 64) + h) * 128 + w];
  }
  for (int i = tid; i < 1024 * 8; i += 256) w1s[i] = w1[i];

  float csum[4] = {0.f, 0.f, 0.f, 0.f};

  for (int hc = 0; hc < 4; ++hc) {
    __syncthreads();
    for (int i = tid; i < 16 * 1024; i += 256) {
      int hl = i >> 10, j = i & 1023;
      const float* xr = xs + (hc * 16 + hl) * 8;
      const bf16* wr = w1s + j * 8;
      float v = b1[j];
#pragma unroll
      for (int c = 0; c < 8; ++c) v += xr[c] * (float)wr[c];
      h1s[hl * TLDH + j] = (bf16)mishf(v);
    }
    __syncthreads();
    v8f acc[4];
#pragma unroll
    for (int f = 0; f < 4; ++f) acc[f] = vzero8();
    for (int kt = 0; kt < 1024; kt += 32) {
      v16bf a = frag_a_rm(h1s + kt, TLDH);
#pragma unroll
      for (int f = 0; f < 4; ++f) {
        int nt = wave + f * 8;
        v16bf bb = frag_b_rm(w2 + (long)(nt * 16) * 1024 + kt, 1024);
        acc[f] = wmma_bf(a, bb, acc[f]);
      }
    }
#pragma unroll
    for (int f = 0; f < 4; ++f) {
      int n = (wave + f * 8) * 16 + (lane & 15);
      float bv = b2[n];
      float s = 0.f;
#pragma unroll
      for (int j = 0; j < 8; ++j) s += mishf(acc[f][j] + bv);
      csum[f] += s;
    }
  }
#pragma unroll
  for (int f = 0; f < 4; ++f) {
    float tot = csum[f] + __shfl_xor(csum[f], 16, 32);
    int n = (wave + f * 8) * 16 + (lane & 15);
    if (lane < 16) {
      float v = tot + pe[w * TD + n];
      srcf[(long)bw * TD + n] = v;
      srcb[(long)bw * TD + n] = (bf16)v;
    }
  }
}

// ======================= fused flash attention (wave32, WMMA) =======================
// One block per (b, head). Addressing: tensor[idx] = base + b*Bs + token*Ls + h*64 + d
#define KLD 72
#define PLD 40
__global__ __launch_bounds__(256)
void attn_kernel(const bf16* __restrict__ Qb, int qB, int qL,
                 const bf16* __restrict__ Kb, int kB, int kL,
                 const bf16* __restrict__ Vb, int vB, int vL,
                 bf16* __restrict__ Ob, int oB, int oL,
                 int Lq, int Lk, float scale)
{
  __shared__ __align__(16) bf16 Kl[128 * KLD];
  __shared__ __align__(16) bf16 Vl[128 * KLD];
  __shared__ __align__(16) bf16 Pl[8 * 16 * PLD];
  const int tid = threadIdx.x, lane = tid & 31, wave = tid >> 5;
  const int b = blockIdx.x >> 3, h = blockIdx.x & 7;
  const int hco = h * 64;
  const int LkP = (Lk + 31) & ~31;

  for (int i = tid; i < LkP * 8; i += 256) {
    int j = i >> 3, d0 = (i & 7) * 8;
    u32x4 kz = {0u,0u,0u,0u}, vz = {0u,0u,0u,0u};
    if (j < Lk) {
      kz = *(const u32x4*)(Kb + (long)b * kB + (long)j * kL + hco + d0);
      vz = *(const u32x4*)(Vb + (long)b * vB + (long)j * vL + hco + d0);
    }
    *(u32x4*)(Kl + j * KLD + d0) = kz;
    *(u32x4*)(Vl + j * KLD + d0) = vz;
  }
  __syncthreads();

  const int nQ = (Lq + 15) >> 4;
  bf16* Pw = Pl + wave * 16 * PLD;
  const int cn = lane & 15, rh = (lane >> 4) * 8;

  for (int qt = wave; qt < nQ; qt += 8) {
    int qrow = qt * 16 + cn; if (qrow >= Lq) qrow = Lq - 1;
    const bf16* qp = Qb + (long)b * qB + (long)qrow * qL + hco + ((lane >> 4) << 3);
    v16bf qa0, qa1;
    {
      v8bf lo = *(const v8bf*)(qp), hi = *(const v8bf*)(qp + 16);
#pragma unroll
      for (int j = 0; j < 8; ++j) { qa0[j] = lo[j]; qa0[j + 8] = hi[j]; }
      lo = *(const v8bf*)(qp + 32); hi = *(const v8bf*)(qp + 48);
#pragma unroll
      for (int j = 0; j < 8; ++j) { qa1[j] = lo[j]; qa1[j + 8] = hi[j]; }
    }

    v8f o0 = vzero8(), o1 = vzero8(), o2 = vzero8(), o3 = vzero8();
    float mrun[8], srun[8];
#pragma unroll
    for (int j = 0; j < 8; ++j) { mrun[j] = -3.0e38f; srun[j] = 0.f; }

    for (int kc = 0; kc < LkP; kc += 32) {
      v8f s0 = vzero8(), s1 = vzero8();
      s0 = wmma_bf(qa0, frag_b_rm(Kl + kc * KLD, KLD), s0);
      s0 = wmma_bf(qa1, frag_b_rm(Kl + kc * KLD + 32, KLD), s0);
      s1 = wmma_bf(qa0, frag_b_rm(Kl + (kc + 16) * KLD, KLD), s1);
      s1 = wmma_bf(qa1, frag_b_rm(Kl + (kc + 16) * KLD + 32, KLD), s1);

      const bool m0 = (kc + cn) >= Lk;
      const bool m1 = (kc + 16 + cn) >= Lk;
      float p0[8], p1[8];
#pragma unroll
      for (int j = 0; j < 8; ++j) {
        float a0 = m0 ? -1.0e30f : s0[j] * scale;
        float a1 = m1 ? -1.0e30f : s1[j] * scale;
        float mx = fmaxf(a0, a1);
        mx = fmaxf(mx, __shfl_xor(mx, 1, 32));
        mx = fmaxf(mx, __shfl_xor(mx, 2, 32));
        mx = fmaxf(mx, __shfl_xor(mx, 4, 32));
        mx = fmaxf(mx, __shfl_xor(mx, 8, 32));
        float mn = fmaxf(mrun[j], mx);
        float fac = expf(mrun[j] - mn);
        mrun[j] = mn;
        float e0 = expf(a0 - mn), e1 = expf(a1 - mn);
        float ps = e0 + e1;
        ps += __shfl_xor(ps, 1, 32);
        ps += __shfl_xor(ps, 2, 32);
        ps += __shfl_xor(ps, 4, 32);
        ps += __shfl_xor(ps, 8, 32);
        srun[j] = srun[j] * fac + ps;
        o0[j] *= fac; o1[j] *= fac; o2[j] *= fac; o3[j] *= fac;
        p0[j] = e0; p1[j] = e1;
      }
#pragma unroll
      for (int j = 0; j < 8; ++j) {
        Pw[(rh + j) * PLD + cn]      = (bf16)p0[j];
        Pw[(rh + j) * PLD + 16 + cn] = (bf16)p1[j];
      }
      asm volatile("s_wait_dscnt 0x0" ::: "memory");
      v16bf pa = frag_a_rm(Pw, PLD);
      o0 = wmma_bf(pa, frag_b_cm(Vl + kc * KLD + 0,  KLD), o0);
      o1 = wmma_bf(pa, frag_b_cm(Vl + kc * KLD + 16, KLD), o1);
      o2 = wmma_bf(pa, frag_b_cm(Vl + kc * KLD + 32, KLD), o2);
      o3 = wmma_bf(pa, frag_b_cm(Vl + kc * KLD + 48, KLD), o3);
    }
#pragma unroll
    for (int j = 0; j < 8; ++j) {
      int q = qt * 16 + rh + j;
      if (q < Lq) {
        float inv = 1.f / srun[j];
        long base = (long)b * oB + (long)q * oL + hco;
        Ob[base + 0  + cn] = (bf16)(o0[j] * inv);
        Ob[base + 16 + cn] = (bf16)(o1[j] * inv);
        Ob[base + 32 + cn] = (bf16)(o2[j] * inv);
        Ob[base + 48 + cn] = (bf16)(o3[j] * inv);
      }
    }
  }
}

// ======================= layernorm over T=512 (one token / block) =======================
__global__ __launch_bounds__(256)
void ln_kernel(const float* __restrict__ Xp, const float* __restrict__ Rp,
               const float* __restrict__ g, const float* __restrict__ bta,
               float* __restrict__ Yf, bf16* __restrict__ Yb)
{
  __shared__ float red[256];
  const int r = blockIdx.x, t = threadIdx.x;
  const long base = (long)r * TD;
  float x0 = Xp[base + 2 * t], x1 = Xp[base + 2 * t + 1];
  if (Rp) { x0 += Rp[base + 2 * t]; x1 += Rp[base + 2 * t + 1]; }
  red[t] = x0 + x1;
  __syncthreads();
  for (int s = 128; s > 0; s >>= 1) { if (t < s) red[t] += red[t + s]; __syncthreads(); }
  float mean = red[0] * (1.f / 512.f);
  __syncthreads();
  float d0 = x0 - mean, d1 = x1 - mean;
  red[t] = d0 * d0 + d1 * d1;
  __syncthreads();
  for (int s = 128; s > 0; s >>= 1) { if (t < s) red[t] += red[t + s]; __syncthreads(); }
  float rstd = rsqrtf(red[0] * (1.f / 512.f) + 1e-5f);
  float y0 = d0 * rstd * g[2 * t] + bta[2 * t];
  float y1 = d1 * rstd * g[2 * t + 1] + bta[2 * t + 1];
  if (Yf) { Yf[base + 2 * t] = y0; Yf[base + 2 * t + 1] = y1; }
  if (Yb) { Yb[base + 2 * t] = (bf16)y0; Yb[base + 2 * t + 1] = (bf16)y1; }
}

// ======================= small elementwise kernels =======================
__global__ void cvt_kernel(const float* __restrict__ in, bf16* __restrict__ out, int n) {
  int i = blockIdx.x * 256 + threadIdx.x;
  if (i < n) out[i] = (bf16)in[i];
}
__global__ void pe_kernel(float* __restrict__ pe) {
  int i = blockIdx.x * 256 + threadIdx.x;
  if (i < LMAX * TD) {
    int l = i >> 9, d = i & 511;
    int i2 = d >> 1;
    float div = expf(-logf(10000.f) * (float)(2 * i2) / 512.f);
    float a = (float)l * div;
    pe[i] = (d & 1) ? cosf(a) : sinf(a);
  }
}
// cur = buf + pe[l]  (l-major rows: r = l*32 + b)
__global__ void addpe_kernel(const float* __restrict__ buf, const float* __restrict__ pe,
                             float* __restrict__ curf, bf16* __restrict__ curb, int n) {
  int i = blockIdx.x * 256 + threadIdx.x;
  if (i < n) {
    int r = i >> 9, t = i & 511;
    float v = buf[i] + pe[(r >> 5) * TD + t];
    curf[i] = v; curb[i] = (bf16)v;
  }
}
// out = x * (scale+1) + shift ; sc rows r = w*32 + b, cols [shift(0..7) scale(8..15)]
__global__ void outscale_kernel(const float* __restrict__ x, const float* __restrict__ sc,
                                float* __restrict__ out) {
  int i = blockIdx.x * 256 + threadIdx.x;
  if (i < 2097152) {
    int w = i & 127;
    int c = (i >> 13) & 7;
    int b = i >> 16;
    int r = w * 32 + b;
    out[i] = x[i] * (sc[r * 16 + 8 + c] + 1.f) + sc[r * 16 + c];
  }
}

// ======================= host orchestration =======================
extern "C" void kernel_launch(void* const* d_in, const int* in_sizes, int n_in,
                              void* d_out, int out_size, void* d_ws, size_t ws_size,
                              hipStream_t stream)
{
  (void)in_sizes; (void)n_in; (void)out_size; (void)ws_size;
  const float* X      = (const float*)d_in[0];
  const float* YENC   = (const float*)d_in[1];
  const float* TTS_W1 = (const float*)d_in[2];
  const float* TTS_B1 = (const float*)d_in[3];
  const float* TTS_W2 = (const float*)d_in[4];
  const float* TTS_B2 = (const float*)d_in[5];
  const float* ST_W1  = (const float*)d_in[6];
  const float* ST_B1  = (const float*)d_in[7];
  const float* ST_W2  = (const float*)d_in[8];
  const float* ST_B2  = (const float*)d_in[9];
  const float* E_INW  = (const float*)d_in[10];
  const float* E_INB  = (const float*)d_in[11];
  const float* E_OUTW = (const float*)d_in[12];
  const float* E_OUTB = (const float*)d_in[13];
  const float* E_F1W  = (const float*)d_in[14];
  const float* E_F1B  = (const float*)d_in[15];
  const float* E_F2W  = (const float*)d_in[16];
  const float* E_F2B  = (const float*)d_in[17];
  const float* E_L1G  = (const float*)d_in[18];
  const float* E_L1B  = (const float*)d_in[19];
  const float* E_L2G  = (const float*)d_in[20];
  const float* E_L2B  = (const float*)d_in[21];
  const float* E_NG   = (const float*)d_in[22];
  const float* E_NB   = (const float*)d_in[23];
  const float* D_SAIW = (const float*)d_in[24];
  const float* D_SAIB = (const float*)d_in[25];
  const float* D_SAOW = (const float*)d_in[26];
  const float* D_SAOB = (const float*)d_in[27];
  const float* D_CAIW = (const float*)d_in[28];
  const float* D_CAIB = (const float*)d_in[29];
  const float* D_CAOW = (const float*)d_in[30];
  const float* D_CAOB = (const float*)d_in[31];
  const float* D_F1W  = (const float*)d_in[32];
  const float* D_F1B  = (const float*)d_in[33];
  const float* D_F2W  = (const float*)d_in[34];
  const float* D_F2B  = (const float*)d_in[35];
  const float* D_L1G  = (const float*)d_in[36];
  const float* D_L1B  = (const float*)d_in[37];
  const float* D_L2G  = (const float*)d_in[38];
  const float* D_L2B  = (const float*)d_in[39];
  const float* D_L3G  = (const float*)d_in[40];
  const float* D_L3B  = (const float*)d_in[41];
  const float* D_NG   = (const float*)d_in[42];
  const float* D_NB   = (const float*)d_in[43];
  const float* TOCH_W = (const float*)d_in[44];
  const float* TOCH_B = (const float*)d_in[45];

  char* wsp = (char*)d_ws;
  auto alloc = [&](size_t bytes) -> char* {
    char* p = wsp; wsp += (bytes + 255) & ~(size_t)255; return p;
  };

  // bf16 weight copies
  bf16* w_tts1 = (bf16*)alloc(1024 * 8 * 2);
  bf16* w_tts2 = (bf16*)alloc(512 * 1024 * 2);
  bf16* w_st1  = (bf16*)alloc(1024 * 128 * 2);
  bf16* w_st2  = (bf16*)alloc(512 * 1024 * 2);
  bf16* w_einw = (bf16*)alloc((size_t)3 * 1536 * 512 * 2);
  bf16* w_eout = (bf16*)alloc((size_t)3 * 512 * 512 * 2);
  bf16* w_ef1  = (bf16*)alloc((size_t)3 * 2048 * 512 * 2);
  bf16* w_ef2  = (bf16*)alloc((size_t)3 * 512 * 2048 * 2);
  bf16* w_sai  = (bf16*)alloc((size_t)3 * 1536 * 512 * 2);
  bf16* w_sao  = (bf16*)alloc((size_t)3 * 512 * 512 * 2);
  bf16* w_cai  = (bf16*)alloc((size_t)3 * 1536 * 512 * 2);
  bf16* w_cao  = (bf16*)alloc((size_t)3 * 512 * 512 * 2);
  bf16* w_df1  = (bf16*)alloc((size_t)3 * 2048 * 512 * 2);
  bf16* w_df2  = (bf16*)alloc((size_t)3 * 512 * 2048 * 2);
  bf16* w_toch = (bf16*)alloc(16 * 512 * 2);

  // activations / scratch
  float* pe    = (float*)alloc((size_t)LMAX * TD * 4);
  float* src_f = (float*)alloc((size_t)4096 * TD * 4);
  bf16*  src_b = (bf16*) alloc((size_t)4096 * TD * 2);
  bf16*  qkv_b = (bf16*) alloc((size_t)4128 * 1536 * 2);
  bf16*  ao_b  = (bf16*) alloc((size_t)4128 * TD * 2);
  float* proj  = (float*)alloc((size_t)4128 * TD * 4);
  bf16*  ff_b  = (bf16*) alloc((size_t)4128 * 2048 * 2);
  bf16*  mem_b = (bf16*) alloc((size_t)4096 * TD * 2);
  bf16*  kv_b  = (bf16*) alloc((size_t)3 * 4096 * 1024 * 2);
  float* buf   = (float*)alloc((size_t)LMAX * 32 * TD * 4);
  float* cur_f = (float*)alloc((size_t)4128 * TD * 4);
  bf16*  cur_b = (bf16*) alloc((size_t)4128 * TD * 2);
  bf16*  q_b   = (bf16*) alloc((size_t)4128 * TD * 2);
  bf16*  seq_b = (bf16*) alloc((size_t)4096 * TD * 2);
  bf16*  y_b   = (bf16*) alloc(32 * 128 * 2);
  bf16*  st_b  = (bf16*) alloc(32 * 1024 * 2);
  float* sc_f  = (float*)alloc((size_t)4096 * 16 * 4);

  auto cvt = [&](const float* s, bf16* d, size_t n) {
    cvt_kernel<<<dim3((unsigned)((n + 255) / 256)), dim3(256), 0, stream>>>(s, d, (int)n);
  };
  auto gemm = [&](const bf16* A, const bf16* W, const float* bias, float* Yf, bf16* Yb,
                  int M, int N, int K, int act) {
    dim3 g((N + GBN - 1) / GBN, (M + GBM - 1) / GBM);
    gemm_kernel<<<g, dim3(256), 0, stream>>>(A, W, bias, Yf, Yb, M, N, K, act);
  };
  auto attn = [&](const bf16* Q, int qB, int qL, const bf16* K, int kB, int kL,
                  const bf16* V, int vB, int vL, bf16* O, int oB, int oL, int Lq, int Lk) {
    attn_kernel<<<dim3(BSZ * NHD), dim3(256), 0, stream>>>(Q, qB, qL, K, kB, kL, V, vB, vL,
                                                           O, oB, oL, Lq, Lk, 0.125f);
  };
  auto ln = [&](const float* Xp, const float* Rp, const float* g, const float* bb,
                float* Yf, bf16* Yb, int M) {
    ln_kernel<<<dim3(M), dim3(256), 0, stream>>>(Xp, Rp, g, bb, Yf, Yb);
  };

  // ---- weight conversion ----
  cvt(TTS_W1, w_tts1, 1024 * 8);
  cvt(TTS_W2, w_tts2, 512 * 1024);
  cvt(ST_W1,  w_st1,  1024 * 128);
  cvt(ST_W2,  w_st2,  512 * 1024);
  cvt(E_INW,  w_einw, (size_t)3 * 1536 * 512);
  cvt(E_OUTW, w_eout, (size_t)3 * 512 * 512);
  cvt(E_F1W,  w_ef1,  (size_t)3 * 2048 * 512);
  cvt(E_F2W,  w_ef2,  (size_t)3 * 512 * 2048);
  cvt(D_SAIW, w_sai,  (size_t)3 * 1536 * 512);
  cvt(D_SAOW, w_sao,  (size_t)3 * 512 * 512);
  cvt(D_CAIW, w_cai,  (size_t)3 * 1536 * 512);
  cvt(D_CAOW, w_cao,  (size_t)3 * 512 * 512);
  cvt(D_F1W,  w_df1,  (size_t)3 * 2048 * 512);
  cvt(D_F2W,  w_df2,  (size_t)3 * 512 * 2048);
  cvt(TOCH_W, w_toch, 16 * 512);
  cvt(YENC,   y_b,    32 * 128);

  pe_kernel<<<dim3((LMAX * TD + 255) / 256), dim3(256), 0, stream>>>(pe);

  // ---- ToTimeSeries + PE -> src ----
  tts_kernel<<<dim3(4096), dim3(256), 0, stream>>>(X, w_tts1, TTS_B1, w_tts2, TTS_B2,
                                                   pe, src_f, src_b);

  // ---- start token MLP -> buf rows 0..31 (l-major: row = l*32+b) ----
  gemm(y_b, w_st1, ST_B1, nullptr, st_b, 32, 1024, 128, 1);
  gemm(st_b, w_st2, ST_B2, buf, nullptr, 32, 512, 1024, 1);

  // ---- encoder (b-major tokens: row = b*128 + w) ----
  for (int l = 0; l < 3; ++l) {
    const bf16* inw = w_einw + (size_t)l * 1536 * 512;
    gemm(src_b, inw, E_INB + l * 1536, nullptr, qkv_b, 4096, 1536, 512, 0);
    attn(qkv_b, 128 * 1536, 1536, qkv_b + 512, 128 * 1536, 1536,
         qkv_b + 1024, 128 * 1536, 1536, ao_b, 128 * 512, 512, 128, 128);
    gemm(ao_b, w_eout + (size_t)l * 512 * 512, E_OUTB + l * 512, proj, nullptr, 4096, 512, 512, 0);
    ln(src_f, proj, E_L1G + l * 512, E_L1B + l * 512, src_f, src_b, 4096);
    gemm(src_b, w_ef1 + (size_t)l * 2048 * 512, E_F1B + l * 2048, nullptr, ff_b, 4096, 2048, 512, 2);
    gemm(ff_b, w_ef2 + (size_t)l * 512 * 2048, E_F2B + l * 512, proj, nullptr, 4096, 512, 2048, 0);
    ln(src_f, proj, E_L2G + l * 512, E_L2B + l * 512, src_f, src_b, 4096);
  }
  ln(src_f, nullptr, E_NG, E_NB, nullptr, mem_b, 4096);

  // ---- hoisted cross-attention K/V (constant across all decode steps) ----
  for (int l = 0; l < 3; ++l) {
    gemm(mem_b, w_cai + (size_t)l * 1536 * 512 + (size_t)512 * 512,
         D_CAIB + l * 1536 + 512, nullptr, kv_b + (size_t)l * 4096 * 1024,
         4096, 1024, 512, 0);
  }

  // ---- autoregressive decode: 128 steps, l-major target buffer ----
  for (int i = 0; i < 128; ++i) {
    const int Lv = i + 1;
    const int Ms = 32 * Lv;
    addpe_kernel<<<dim3((Ms * TD + 255) / 256), dim3(256), 0, stream>>>(buf, pe, cur_f, cur_b,
                                                                        Ms * TD);
    for (int l = 0; l < 3; ++l) {
      // self attention
      gemm(cur_b, w_sai + (size_t)l * 1536 * 512, D_SAIB + l * 1536, nullptr, qkv_b,
           Ms, 1536, 512, 0);
      attn(qkv_b, 1536, 32 * 1536, qkv_b + 512, 1536, 32 * 1536,
           qkv_b + 1024, 1536, 32 * 1536, ao_b, 512, 32 * 512, Lv, Lv);
      gemm(ao_b, w_sao + (size_t)l * 512 * 512, D_SAOB + l * 512, proj, nullptr, Ms, 512, 512, 0);
      ln(cur_f, proj, D_L1G + l * 512, D_L1B + l * 512, cur_f, cur_b, Ms);
      // cross attention (K/V hoisted)
      gemm(cur_b, w_cai + (size_t)l * 1536 * 512, D_CAIB + l * 1536, nullptr, q_b, Ms, 512, 512, 0);
      attn(q_b, 512, 32 * 512, kv_b + (size_t)l * 4096 * 1024, 128 * 1024, 1024,
           kv_b + (size_t)l * 4096 * 1024 + 512, 128 * 1024, 1024,
           ao_b, 512, 32 * 512, Lv, 128);
      gemm(ao_b, w_cao + (size_t)l * 512 * 512, D_CAOB + l * 512, proj, nullptr, Ms, 512, 512, 0);
      ln(cur_f, proj, D_L2G + l * 512, D_L2B + l * 512, cur_f, cur_b, Ms);
      // ffn
      gemm(cur_b, w_df1 + (size_t)l * 2048 * 512, D_F1B + l * 2048, nullptr, ff_b, Ms, 2048, 512, 2);
      gemm(ff_b, w_df2 + (size_t)l * 512 * 2048, D_F2B + l * 512, proj, nullptr, Ms, 512, 2048, 0);
      ln(cur_f, proj, D_L3G + l * 512, D_L3B + l * 512, cur_f, cur_b, Ms);
    }
    // final decoder norm on the 32 last-position tokens only; append to buf + record seq
    ln(cur_f + (size_t)i * 32 * TD, nullptr, D_NG, D_NB,
       buf + (size_t)(i + 1) * 32 * TD, seq_b + (size_t)i * 32 * TD, 32);
  }

  // ---- to_channels + final scale/shift ----
  gemm(seq_b, w_toch, TOCH_B, sc_f, nullptr, 4096, 16, 512, 0);
  outscale_kernel<<<dim3(8192), dim3(256), 0, stream>>>(X, sc_f, (float*)d_out);
}
